// VQLayer_28046136443276
// MI455X (gfx1250) — matrix operand order repaired and checked by reading, not compiled, not run on hardware
//
#include <hip/hip_runtime.h>
#include <hip/hip_bf16.h>

// ---------------------------------------------------------------------------
// VQ layer for MI455X (gfx1250), wave32 + WMMA bf16 + async global->LDS.
// Sizes: B=4, L=2048 (NTOK=8192), D_MODEL=1024, VQ_DIM=64, K_CODES=8192.
// Never materializes the 8192x8192 distance matrix; two fused WMMA passes
// (token-major argmin, code-major top-5).  The streamed operand of each pass
// is double-buffered through LDS with global_load_async_to_lds_b128 so 8
// waves/WGP share one L2 read stream instead of eight.
// ---------------------------------------------------------------------------

#define NTOK    8192
#define DMODEL  1024
#define VDIM    64
#define KCODES  8192
#define LSEQ    2048
#define NB      4

#define TPAD    72      // padded LDS row stride in halfs (144 B, bank-conflict free)

typedef __bf16        v16bf __attribute__((ext_vector_type(16)));
typedef float         v8f   __attribute__((ext_vector_type(8)));
typedef unsigned int  v8u   __attribute__((ext_vector_type(8)));

// Native f32->bf16 (RNE): let the backend select v_cvt_pk_bf16_f32-class ops
// instead of a manual 5-instruction rounding chain.
__device__ __forceinline__ unsigned short f2bf(float f) {
    __bf16 x = (__bf16)f;
    return __builtin_bit_cast(unsigned short, x);
}
__device__ __forceinline__ unsigned int pack2(float a, float b) {
    __bf16 x = (__bf16)a, y = (__bf16)b;
    return (unsigned int)__builtin_bit_cast(unsigned short, x) |
           ((unsigned int)__builtin_bit_cast(unsigned short, y) << 16);
}
__device__ __forceinline__ float huber(float d) {
    float ad = fabsf(d);
    return (ad < 1.0f) ? 0.5f * d * d : ad - 0.5f;
}

__device__ __forceinline__ void wait_async0() {
#if defined(__has_builtin) && __has_builtin(__builtin_amdgcn_s_wait_asynccnt)
    __builtin_amdgcn_s_wait_asynccnt(0);
#else
    asm volatile("s_wait_asynccnt 0x0" ::: "memory");
#endif
}

// One 16-byte async chunk: global -> LDS (tracked by ASYNCcnt).
__device__ __forceinline__ void async_cp16(unsigned int lds_byte_off, const void* gaddr) {
    asm volatile("global_load_async_to_lds_b128 %0, %1, off"
                 :: "v"(lds_byte_off), "v"(gaddr)
                 : "memory");
}

// Stage 64 rows x 64 halfs (128 B each) from src[(rowBase+r)*64] into
// dst[r*TPAD], as 512 x 16B chunks spread over the block.
__device__ __forceinline__ void stage_tile64(unsigned short* dst_shared,
                                             const unsigned short* __restrict__ src,
                                             int rowBase, int tid, int nthreads) {
    for (int idx = tid; idx < 512; idx += nthreads) {
        int r = idx >> 3, c = idx & 7;                       // chunk c = halfs [8c, 8c+8)
        unsigned int lds = (unsigned int)(uintptr_t)(dst_shared + r * TPAD + c * 8);
        const void* g = (const void*)(src + (rowBase + r) * VDIM + c * 8);
        async_cp16(lds, g);
    }
}

// A-fragment (16x32 bf16): lane: two contiguous 16B chunks at k0+16*half, k0+16*half+8.
__device__ __forceinline__ v16bf load_a_bf16(const unsigned short* __restrict__ base,
                                             int ld, int row, int k0, int lhalf) {
    const unsigned short* p = base + row * ld + k0 + 16 * lhalf;
    uint4 c0 = *(const uint4*)p;
    uint4 c1 = *(const uint4*)(p + 8);
    v8u r;
    r[0] = c0.x; r[1] = c0.y; r[2] = c0.z; r[3] = c0.w;
    r[4] = c1.x; r[5] = c1.y; r[6] = c1.z; r[7] = c1.w;
    return __builtin_bit_cast(v16bf, r);
}

// A-fragment from LDS tile (padded stride TPAD).
__device__ __forceinline__ v16bf load_a_lds(const unsigned short* buf,
                                            int rowLocal, int k0, int lhalf) {
    const unsigned short* p = buf + rowLocal * TPAD + k0 + 16 * lhalf;
    uint4 c0 = *(const uint4*)p;
    uint4 c1 = *(const uint4*)(p + 8);
    v8u r;
    r[0] = c0.x; r[1] = c0.y; r[2] = c0.z; r[3] = c0.w;
    r[4] = c1.x; r[5] = c1.y; r[6] = c1.z; r[7] = c1.w;
    return __builtin_bit_cast(v16bf, r);
}

// A-fragment sourced from fp32 (converted on the fly, native pk converts).
__device__ __forceinline__ v16bf load_a_f32(const float* __restrict__ base,
                                            int ld, int row, int k0, int lhalf) {
    const float* p = base + row * ld + k0 + 16 * lhalf;
    float4 f0 = *(const float4*)p;
    float4 f1 = *(const float4*)(p + 4);
    float4 g0 = *(const float4*)(p + 8);
    float4 g1 = *(const float4*)(p + 12);
    v8u r;
    r[0] = pack2(f0.x, f0.y); r[1] = pack2(f0.z, f0.w);
    r[2] = pack2(f1.x, f1.y); r[3] = pack2(f1.z, f1.w);
    r[4] = pack2(g0.x, g0.y); r[5] = pack2(g0.z, g0.w);
    r[6] = pack2(g1.x, g1.y); r[7] = pack2(g1.z, g1.w);
    return __builtin_bit_cast(v16bf, r);
}

// B-fragment (32x16 bf16): lane reads 32 contiguous bytes at col*ld + k0 + 16*half.
__device__ __forceinline__ v16bf load_b_bf16(const unsigned short* __restrict__ base,
                                             int ld, int col, int k0, int lhalf) {
    const unsigned short* p = base + col * ld + k0 + 16 * lhalf;
    return __builtin_bit_cast(v16bf, *(const v8u*)p);
}

// B-fragment from LDS tile (padded stride TPAD); two 16B loads (16B alignment).
__device__ __forceinline__ v16bf load_b_lds(const unsigned short* buf,
                                            int colLocal, int k0, int lhalf) {
    const unsigned short* p = buf + colLocal * TPAD + k0 + 16 * lhalf;
    uint4 c0 = *(const uint4*)p;
    uint4 c1 = *(const uint4*)(p + 8);
    v8u r;
    r[0] = c0.x; r[1] = c0.y; r[2] = c0.z; r[3] = c0.w;
    r[4] = c1.x; r[5] = c1.y; r[6] = c1.z; r[7] = c1.w;
    return __builtin_bit_cast(v16bf, r);
}

#define WMMA_BF16(A, B, C) \
    __builtin_amdgcn_wmma_f32_16x16x32_bf16(false, (A), false, (B), (short)0, (C), false, false)

// ---------------------------------------------------------------------------
// Prep kernels.
// ---------------------------------------------------------------------------
__global__ __launch_bounds__(256) void k_prep_w(const float* __restrict__ pw,
                                                const float* __restrict__ piw,
                                                unsigned short* __restrict__ wbf_t,
                                                unsigned short* __restrict__ wibf_t) {
    int i = blockIdx.x * blockDim.x + threadIdx.x;
    if (i < DMODEL * VDIM) {
        int k = i >> 6, n = i & 63;            // pw[k*64+n]
        wbf_t[n * DMODEL + k] = f2bf(pw[i]);
        int k2 = i >> 10, n2 = i & 1023;       // piw[k2*1024+n2]
        wibf_t[n2 * VDIM + k2] = f2bf(piw[i]);
    }
}

__global__ __launch_bounds__(256) void k_prep_e(const float* __restrict__ emb,
                                                float* __restrict__ E_f,
                                                unsigned short* __restrict__ E_bf) {
    int lane = threadIdx.x & 31, wave = threadIdx.x >> 5;
    int row = blockIdx.x * 8 + wave;
    int d0 = lane * 2;
    float v0 = emb[row * VDIM + d0], v1 = emb[row * VDIM + d0 + 1];
    float ss = v0 * v0 + v1 * v1;
    #pragma unroll
    for (int off = 1; off < 32; off <<= 1) ss += __shfl_xor(ss, off, 32);
    float inv = rsqrtf(ss);
    v0 *= inv; v1 *= inv;
    E_f[row * VDIM + d0]     = v0;
    E_f[row * VDIM + d0 + 1] = v1;
    ((unsigned int*)E_bf)[row * 32 + lane] = pack2(v0, v1);
}

__global__ __launch_bounds__(256) void k_prep_misc(const float* __restrict__ attn,
                                                   const float* __restrict__ count,
                                                   float* __restrict__ msum,
                                                   float* __restrict__ loss_acc,
                                                   float* __restrict__ out_count) {
    __shared__ float sm[NB];
    int tid = threadIdx.x;
    if (tid < NB) sm[tid] = 0.0f;
    __syncthreads();
    int b = tid >> 6, i = tid & 63;
    float s = 0.0f;
    for (int j = 0; j < 32; ++j) s += attn[b * LSEQ + i * 32 + j];
    atomicAdd(&sm[b], s);
    __syncthreads();
    if (tid < NB) msum[tid] = sm[tid];
    if (tid == 4) loss_acc[0] = 0.0f;
    for (int k = tid; k < KCODES; k += 256) out_count[k] = count[k];
}

// ---------------------------------------------------------------------------
// hp = normalize(h @ proj_w + b); writes fp32 + bf16.  One wave = 16 rows x 64.
// ---------------------------------------------------------------------------
__global__ __launch_bounds__(256) void k_proj(const float* __restrict__ h,
                                              const float* __restrict__ pb,
                                              const unsigned short* __restrict__ wbf_t,
                                              float* __restrict__ hp_f,
                                              unsigned short* __restrict__ hp_bf) {
    int lane = threadIdx.x & 31, wave = threadIdx.x >> 5;
    int l15 = lane & 15, lhalf = lane >> 4;
    int rowBase = (blockIdx.x * 8 + wave) * 16;

    v8f acc[4] = {};
    for (int k0 = 0; k0 < DMODEL; k0 += 32) {
        v16bf a = load_a_f32(h, DMODEL, rowBase + l15, k0, lhalf);
        #pragma unroll
        for (int t = 0; t < 4; ++t) {
            v16bf b = load_b_bf16(wbf_t, DMODEL, t * 16 + l15, k0, lhalf);
            acc[t] = WMMA_BF16(a, b, acc[t]);
        }
    }
    float ss[8] = {};
    #pragma unroll
    for (int t = 0; t < 4; ++t) {
        float bias = pb[t * 16 + l15];
        #pragma unroll
        for (int r = 0; r < 8; ++r) {
            float v = acc[t][r] + bias;
            acc[t][r] = v;
            ss[r] += v * v;
        }
    }
    #pragma unroll
    for (int r = 0; r < 8; ++r) {
        float s = ss[r];
        #pragma unroll
        for (int off = 1; off < 16; off <<= 1) s += __shfl_xor(s, off, 32);
        ss[r] = rsqrtf(s);
    }
    #pragma unroll
    for (int t = 0; t < 4; ++t) {
        int n = t * 16 + l15;
        #pragma unroll
        for (int r = 0; r < 8; ++r) {
            int m = rowBase + r + 8 * lhalf;
            float v = acc[t][r] * ss[r];
            hp_f[m * VDIM + n]  = v;
            hp_bf[m * VDIM + n] = f2bf(v);
        }
    }
}

// ---------------------------------------------------------------------------
// Token-major pass: argmin over all codes.  E is double-buffered through LDS
// with async copies (64 codes / 8 KB per tile), shared by all 8 waves.
// ---------------------------------------------------------------------------
__global__ __launch_bounds__(256) void k_assign(const unsigned short* __restrict__ hp_bf,
                                                const unsigned short* __restrict__ E_bf,
                                                const float* __restrict__ attn,
                                                int* __restrict__ code_i,
                                                float* __restrict__ out_code) {
    __shared__ unsigned short sE[2][64 * TPAD];
    int tid = threadIdx.x;
    int lane = tid & 31, wave = tid >> 5;
    int l15 = lane & 15, lhalf = lane >> 4;
    int rowBase = (blockIdx.x * 8 + wave) * 16;

    v16bf a0 = load_a_bf16(hp_bf, VDIM, rowBase + l15, 0,  lhalf);
    v16bf a1 = load_a_bf16(hp_bf, VDIM, rowBase + l15, 32, lhalf);

    float best[8]; int bidx[8];
    #pragma unroll
    for (int r = 0; r < 8; ++r) { best[r] = 3.0e38f; bidx[r] = 0; }

    const int NT = KCODES / 64;                  // 128 staged tiles
    stage_tile64(sE[0], E_bf, 0, tid, 256);
    for (int t = 0; t < NT; ++t) {
        wait_async0();
        __syncthreads();                         // tile t resident in LDS
        if (t + 1 < NT) stage_tile64(sE[(t + 1) & 1], E_bf, (t + 1) * 64, tid, 256);
        const unsigned short* buf = sE[t & 1];
        #pragma unroll
        for (int s = 0; s < 4; ++s) {
            v16bf b0 = load_b_lds(buf, s * 16 + l15, 0,  lhalf);
            v16bf b1 = load_b_lds(buf, s * 16 + l15, 32, lhalf);
            v8f acc = {};
            acc = WMMA_BF16(a0, b0, acc);
            acc = WMMA_BF16(a1, b1, acc);
            int n = t * 64 + s * 16 + l15;
            #pragma unroll
            for (int r = 0; r < 8; ++r) {
                float d = 2.0f - 2.0f * acc[r];
                if (d < best[r]) { best[r] = d; bidx[r] = n; }
            }
        }
        __syncthreads();                         // reads done before buffer reuse
    }
    #pragma unroll
    for (int r = 0; r < 8; ++r) {                // argmin across the 16-lane group
        float b = best[r]; int i = bidx[r];
        #pragma unroll
        for (int off = 1; off < 16; off <<= 1) {
            float ob = __shfl_xor(b, off, 32);
            int   oi = __shfl_xor(i, off, 32);
            if (ob < b || (ob == b && oi < i)) { b = ob; i = oi; }
        }
        if (l15 == 0) {
            int m = rowBase + r + 8 * lhalf;
            int c = (attn[m] != 0.0f) ? i : 0;
            code_i[m]   = c;
            out_code[m] = (float)c;
        }
    }
}

// ---------------------------------------------------------------------------
// Code-major pass: per-code top-5 nearest tokens, then EMA codebook update.
// hp is double-buffered through LDS with async copies (64 tokens per tile).
// ---------------------------------------------------------------------------
__global__ __launch_bounds__(128) void k_ema(const unsigned short* __restrict__ hp_bf,
                                             const unsigned short* __restrict__ E_bf,
                                             const float* __restrict__ hp_f,
                                             const float* __restrict__ E_f,
                                             const float* __restrict__ attn,
                                             float* __restrict__ out_emb) {
    __shared__ unsigned short sH[2][64 * TPAD];
    __shared__ int sm[4][16 * 6];
    int tid = threadIdx.x;
    int lane = tid & 31, wave = tid >> 5;
    int l15 = lane & 15, lhalf = lane >> 4;
    int codeBase = (blockIdx.x * 4 + wave) * 16;

    v16bf b0 = load_b_bf16(E_bf, VDIM, codeBase + l15, 0,  lhalf);
    v16bf b1 = load_b_bf16(E_bf, VDIM, codeBase + l15, 32, lhalf);

    float td[5]; int ti[5];
    #pragma unroll
    for (int j = 0; j < 5; ++j) { td[j] = 3.0e38f; ti[j] = 0; }

    const int NT = NTOK / 64;                    // 128 staged tiles
    stage_tile64(sH[0], hp_bf, 0, tid, 128);
    for (int t = 0; t < NT; ++t) {
        wait_async0();
        __syncthreads();
        if (t + 1 < NT) stage_tile64(sH[(t + 1) & 1], hp_bf, (t + 1) * 64, tid, 128);
        const unsigned short* buf = sH[t & 1];
        #pragma unroll
        for (int s = 0; s < 4; ++s) {
            int tokBase = t * 64 + s * 16;
            v16bf a0 = load_a_lds(buf, s * 16 + l15, 0,  lhalf);
            v16bf a1 = load_a_lds(buf, s * 16 + l15, 32, lhalf);
            v8f acc = {};
            acc = WMMA_BF16(a0, b0, acc);
            acc = WMMA_BF16(a1, b1, acc);
            const float4* mp = (const float4*)(attn + tokBase + 8 * lhalf);
            float4 m0 = mp[0], m1 = mp[1];
            float mk[8] = {m0.x, m0.y, m0.z, m0.w, m1.x, m1.y, m1.z, m1.w};
            #pragma unroll
            for (int r = 0; r < 8; ++r) {
                float d = 2.0f - 2.0f * acc[r];
                d = (mk[r] != 0.0f) ? d : 1.0e7f;    // masked tokens excluded
                int tok = tokBase + r + 8 * lhalf;
                if (d < td[4]) {
                    td[4] = d; ti[4] = tok;
                    #pragma unroll
                    for (int j = 4; j > 0; --j) {
                        if (td[j] < td[j - 1]) {
                            float x = td[j]; td[j] = td[j - 1]; td[j - 1] = x;
                            int   y = ti[j]; ti[j] = ti[j - 1]; ti[j - 1] = y;
                        }
                    }
                }
            }
        }
        __syncthreads();
    }
    // merge this lane's top-5 with lane^16 (other row half, same code)
    float od[5]; int oi[5];
    #pragma unroll
    for (int j = 0; j < 5; ++j) {
        od[j] = __shfl_xor(td[j], 16, 32);
        oi[j] = __shfl_xor(ti[j], 16, 32);
    }
    float md[5]; int mi[5];
    int p = 0, q = 0;
    #pragma unroll
    for (int j = 0; j < 5; ++j) {
        bool takeA = (td[p] < od[q]) || (td[p] == od[q] && ti[p] <= oi[q]);
        if (takeA) { md[j] = td[p]; mi[j] = ti[p]; ++p; }
        else       { md[j] = od[q]; mi[j] = oi[q]; ++q; }
    }
    if (lhalf == 0) {
        #pragma unroll
        for (int j = 0; j < 5; ++j) sm[wave][l15 * 6 + j] = mi[j];
        sm[wave][l15 * 6 + 5] = (md[0] > 0.1f) ? 1 : 0;   // min distance > 0.1
    }
    __syncthreads();
    int d0 = lane * 2;
    for (int c = 0; c < 16; ++c) {
        int code = codeBase + c;
        int flag = sm[wave][c * 6 + 5];
        int i0 = sm[wave][c * 6 + 0], i1 = sm[wave][c * 6 + 1], i2 = sm[wave][c * 6 + 2];
        int i3 = sm[wave][c * 6 + 3], i4 = sm[wave][c * 6 + 4];
        #pragma unroll
        for (int e = 0; e < 2; ++e) {
            int d = d0 + e;
            float base = E_f[code * VDIM + d];
            float v = base;
            if (flag) {
                float hf = 0.2f * (hp_f[i0 * VDIM + d] + hp_f[i1 * VDIM + d] +
                                   hp_f[i2 * VDIM + d] + hp_f[i3 * VDIM + d] +
                                   hp_f[i4 * VDIM + d]);
                v = 0.995f * base + 0.005f * hf;
            }
            out_emb[code * VDIM + d] = v;
        }
    }
}

// ---------------------------------------------------------------------------
// Gather quantized rows (bf16), accumulate vq_loss, scatter counts.
// ---------------------------------------------------------------------------
__global__ __launch_bounds__(256) void k_gather_loss(const int* __restrict__ code_i,
                                                     const float* __restrict__ hp_f,
                                                     const float* __restrict__ E_f,
                                                     const float* __restrict__ attn,
                                                     const float* __restrict__ msum,
                                                     unsigned short* __restrict__ q_bf,
                                                     float* __restrict__ loss_acc,
                                                     float* __restrict__ out_count) {
    int lane = threadIdx.x & 31, wave = threadIdx.x >> 5;
    int tok = blockIdx.x * 8 + wave;
    int c = code_i[tok];
    int d0 = lane * 2;
    float h0 = hp_f[tok * VDIM + d0], h1 = hp_f[tok * VDIM + d0 + 1];
    float e0 = E_f[c * VDIM + d0],    e1 = E_f[c * VDIM + d0 + 1];
    // forward value of huber(sg(hp), q) + 0.25*huber(hp, sg(q)) = 1.25*huber(hp, q)
    float l = 1.25f * (huber(h0 - e0) + huber(h1 - e1));
    ((unsigned int*)q_bf)[tok * 32 + lane] = pack2(e0, e1);
    #pragma unroll
    for (int off = 1; off < 32; off <<= 1) l += __shfl_xor(l, off, 32);
    if (lane == 0) {
        float m = attn[tok];
        float ms = msum[tok >> 11];
        if (ms > 0.0f && m != 0.0f)
            atomicAdd(loss_acc, l * m / ((float)(NB * VDIM) * ms));
        atomicAdd(out_count + c, 1.0f);
    }
}

// ---------------------------------------------------------------------------
// q_out = quantized @ proj_inv_w + b.  One wave = 16 rows x 1024 cols.
// ---------------------------------------------------------------------------
__global__ __launch_bounds__(256) void k_qout(const unsigned short* __restrict__ q_bf,
                                              const unsigned short* __restrict__ wibf_t,
                                              const float* __restrict__ pib,
                                              float* __restrict__ out_q) {
    int lane = threadIdx.x & 31, wave = threadIdx.x >> 5;
    int l15 = lane & 15, lhalf = lane >> 4;
    int rowBase = (blockIdx.x * 8 + wave) * 16;

    v16bf a0 = load_a_bf16(q_bf, VDIM, rowBase + l15, 0,  lhalf);
    v16bf a1 = load_a_bf16(q_bf, VDIM, rowBase + l15, 32, lhalf);

    for (int nb = 0; nb < DMODEL; nb += 16) {
        v16bf b0 = load_b_bf16(wibf_t, VDIM, nb + l15, 0,  lhalf);
        v16bf b1 = load_b_bf16(wibf_t, VDIM, nb + l15, 32, lhalf);
        v8f acc = {};
        acc = WMMA_BF16(a0, b0, acc);
        acc = WMMA_BF16(a1, b1, acc);
        int n = nb + l15;
        float bias = pib[n];
        #pragma unroll
        for (int r = 0; r < 8; ++r) {
            int m = rowBase + r + 8 * lhalf;
            out_q[m * DMODEL + n] = acc[r] + bias;
        }
    }
}

__global__ void k_finalize(const float* __restrict__ loss_acc, float* __restrict__ out_loss) {
    out_loss[0] = loss_acc[0];
}

// ---------------------------------------------------------------------------
extern "C" void kernel_launch(void* const* d_in, const int* in_sizes, int n_in,
                              void* d_out, int out_size, void* d_ws, size_t ws_size,
                              hipStream_t stream) {
    const float* h    = (const float*)d_in[0];
    const float* attn = (const float*)d_in[1];
    const float* pw   = (const float*)d_in[2];
    const float* pb   = (const float*)d_in[3];
    const float* piw  = (const float*)d_in[4];
    const float* pib  = (const float*)d_in[5];
    const float* emb  = (const float*)d_in[6];
    const float* cnt  = (const float*)d_in[7];

    char* ws = (char*)d_ws;
    float*          hp_f   = (float*)(ws + 0);                 // 2 MB
    float*          E_f    = (float*)(ws + 2097152);           // 2 MB
    unsigned short* hp_bf  = (unsigned short*)(ws + 4194304);  // 1 MB
    unsigned short* E_bf   = (unsigned short*)(ws + 5242880);  // 1 MB
    unsigned short* q_bf   = (unsigned short*)(ws + 6291456);  // 1 MB
    unsigned short* wbf_t  = (unsigned short*)(ws + 7340032);  // 128 KB
    unsigned short* wibf_t = (unsigned short*)(ws + 7471104);  // 128 KB
    int*            code_i = (int*)(ws + 7602176);             // 32 KB
    float*          msum   = (float*)(ws + 7634944);
    float*          lossA  = (float*)(ws + 7634960);

    float* out       = (float*)d_out;
    float* out_q     = out;                                      // 8192*1024
    float* out_code  = out + (size_t)NTOK * DMODEL;              // 8192
    float* out_loss  = out_code + NTOK;                          // 1
    float* out_emb   = out_loss + 1;                             // 8192*64
    float* out_count = out_emb + (size_t)KCODES * VDIM;          // 8192

    k_prep_w     <<<256, 256, 0, stream>>>(pw, piw, wbf_t, wibf_t);
    k_prep_e     <<<KCODES / 8, 256, 0, stream>>>(emb, E_f, E_bf);
    k_prep_misc  <<<1, 256, 0, stream>>>(attn, cnt, msum, lossA, out_count);
    k_proj       <<<NTOK / 128, 256, 0, stream>>>(h, pb, wbf_t, hp_f, hp_bf);
    k_assign     <<<NTOK / 128, 256, 0, stream>>>(hp_bf, E_bf, attn, code_i, out_code);
    k_ema        <<<KCODES / 64, 128, 0, stream>>>(hp_bf, E_bf, hp_f, E_f, attn, out_emb);
    k_gather_loss<<<NTOK / 8, 256, 0, stream>>>(code_i, hp_f, E_f, attn, msum, q_bf, lossA, out_count);
    k_qout       <<<NTOK / 128, 256, 0, stream>>>(q_bf, wibf_t, pib, out_q);
    k_finalize   <<<1, 1, 0, stream>>>(lossA, out_loss);
}
